// CrossAttention_53102975647928
// MI455X (gfx1250) — matrix operand order repaired and checked by reading, not compiled, not run on hardware
//
#include <hip/hip_runtime.h>
#include <hip/hip_bf16.h>
#include <math.h>

#define D_    1024
#define H_    16
#define HD_   64
#define MLP_  4096
#define B_    4
#define M_    1024
#define N_    2048

typedef __bf16 bf16_t;
typedef __attribute__((ext_vector_type(16))) __bf16 v16bf;
typedef __attribute__((ext_vector_type(8)))  __bf16 bf16x8;
typedef __attribute__((ext_vector_type(8)))  float  v8f;

// ---------------- epilogue selectors ----------------
#define EPI_F32_BIAS      0
#define EPI_BF16_BIAS     1
#define EPI_SCALE_F32     2
#define EPI_BF16_TRANS    3
#define EPI_F32_BIAS_ADD  4
#define EPI_GELU_BF16     5

// ---------------- WMMA fragment loaders ----------------
// A operand: 16x32 bf16 tile, rows m..m+15 of a row-major [R,lda] matrix.
// ISA layout (16-bit A 16x32): lane<16 holds M=lane, halves[0..7]=K k0..k0+7,
// halves[8..15]=K k0+16..k0+23; lane>=16 holds M=lane-16 with K offset +8.
__device__ inline v16bf load_a_bf(const bf16_t* __restrict__ A, int lda,
                                  int m0, int k0, int lane) {
  int hi = lane >> 4;
  const bf16_t* r = A + (size_t)(m0 + (lane & 15)) * lda + k0 + hi * 8;
  union { v16bf v; bf16x8 h[2]; } u;
  u.h[0] = *(const bf16x8*)(r);
  u.h[1] = *(const bf16x8*)(r + 16);
  return u.v;
}

// Same tile but source is f32 (attention probabilities) -> convert inline.
__device__ inline v16bf load_a_f32(const float* __restrict__ A, int lda,
                                   int m0, int k0, int lane) {
  int hi = lane >> 4;
  const float* r = A + (size_t)(m0 + (lane & 15)) * lda + k0 + hi * 8;
  union { v16bf v; __bf16 e[16]; } u;
#pragma unroll
  for (int t = 0; t < 8; ++t) u.e[t] = (__bf16)r[t];
#pragma unroll
  for (int t = 0; t < 8; ++t) u.e[8 + t] = (__bf16)r[16 + t];
  return u.v;
}

// B operand: 32x16 (KxN) tile where column n of B is row n of a row-major
// weight matrix W[C,ldb] (computes A·Wᵀ). ISA layout: lane<16 holds column
// N=lane with halves[0..7]=K k0..k0+7, halves[8..15]=K k0+16..k0+23;
// lane>=16 holds column N=lane-16 with K offset +8.
__device__ inline v16bf load_b_bf(const bf16_t* __restrict__ W, int ldb,
                                  int n0, int k0, int lane) {
  const bf16_t* r = W + (size_t)(n0 + (lane & 15)) * ldb + k0 + ((lane >> 4) << 3);
  union { v16bf v; bf16x8 h[2]; } u;
  u.h[0] = *(const bf16x8*)(r);
  u.h[1] = *(const bf16x8*)(r + 16);
  return u.v;
}

// ---------------- generic WMMA GEMM ----------------
// Out[m,n] = epilogue( sum_k A[m,k] * W[n,k] )
// Block = 128 threads = 4 waves arranged 2x2; each wave owns a 64x64 tile.
// Batch dim z decomposed as (b = z/nH, h = z%nH) with independent strides so
// the same kernel serves per-head attention GEMMs and flat GEMMs.
template <int EPI, bool AF32>
__global__ __launch_bounds__(128) void gemm_wmma(
    const void* __restrict__ Abase, size_t sAb, size_t sAh, int lda,
    const bf16_t* __restrict__ Bbase, size_t sBb, size_t sBh, int ldb,
    void* __restrict__ Obase, size_t sOb, size_t sOh, int ldo,
    const float* __restrict__ bias,
    const float* __restrict__ addBase, size_t sAddb,
    int R, int C, int K, int nH, float alpha) {
  int z  = blockIdx.z;
  int bb = z / nH;
  int hh = z - bb * nH;

  size_t aOff = (size_t)bb * sAb + (size_t)hh * sAh;
  const bf16_t* A  = (const bf16_t*)Abase + (AF32 ? 0 : aOff);
  const float*  Af = (const float*)Abase + (AF32 ? aOff : 0);
  const bf16_t* W  = Bbase + (size_t)bb * sBb + (size_t)hh * sBh;

  int lane = threadIdx.x & 31;
  int wv   = threadIdx.x >> 5;
  int m0 = blockIdx.x * 128 + (wv & 1) * 64;
  int n0 = blockIdx.y * 128 + (wv >> 1) * 64;
  if (m0 >= R || n0 >= C) return;

  v8f acc[4][4] = {};

  for (int k0 = 0; k0 < K; k0 += 32) {
    v16bf af[4], bf[4];
#pragma unroll
    for (int i = 0; i < 4; ++i) {
      if constexpr (AF32) af[i] = load_a_f32(Af, lda, m0 + 16 * i, k0, lane);
      else                af[i] = load_a_bf (A,  lda, m0 + 16 * i, k0, lane);
    }
#pragma unroll
    for (int j = 0; j < 4; ++j) bf[j] = load_b_bf(W, ldb, n0 + 16 * j, k0, lane);
#pragma unroll
    for (int i = 0; i < 4; ++i)
#pragma unroll
      for (int j = 0; j < 4; ++j)
        acc[i][j] = __builtin_amdgcn_wmma_f32_16x16x32_bf16(
            false, af[i], false, bf[j], (short)0, acc[i][j], false, false);
  }

  // epilogue: C/D layout -> VGPR r, lane L: (M = r + (L>=16)*8, N = L&15)
  float*  Of = (float*)Obase;
  bf16_t* Ob = (bf16_t*)Obase;
  size_t oOff = (size_t)bb * sOb + (size_t)hh * sOh;
  const float* addp = addBase ? (addBase + (size_t)bb * sAddb) : addBase;
  int rlo = (lane >> 4) << 3;
  int nc  = lane & 15;

#pragma unroll
  for (int i = 0; i < 4; ++i) {
#pragma unroll
    for (int j = 0; j < 4; ++j) {
      int nn = n0 + 16 * j + nc;
      float bv = (bias != nullptr) ? bias[nn] : 0.0f;
#pragma unroll
      for (int r = 0; r < 8; ++r) {
        int mm = m0 + 16 * i + r + rlo;
        float v = acc[i][j][r];
        if constexpr (EPI == EPI_F32_BIAS) {
          Of[oOff + (size_t)mm * ldo + nn] = v + bv;
        } else if constexpr (EPI == EPI_BF16_BIAS) {
          Ob[oOff + (size_t)mm * ldo + nn] = (bf16_t)(v + bv);
        } else if constexpr (EPI == EPI_SCALE_F32) {
          Of[oOff + (size_t)mm * ldo + nn] = v * alpha;
        } else if constexpr (EPI == EPI_BF16_TRANS) {
          Ob[oOff + (size_t)nn * ldo + mm] = (bf16_t)(v + bv);
        } else if constexpr (EPI == EPI_F32_BIAS_ADD) {
          Of[oOff + (size_t)mm * ldo + nn] = v + bv + addp[(size_t)mm * ldo + nn];
        } else if constexpr (EPI == EPI_GELU_BF16) {
          float t = v + bv;
          Ob[oOff + (size_t)mm * ldo + nn] =
              (bf16_t)(0.5f * t * (1.0f + erff(t * 0.70710678118654752f)));
        }
      }
    }
  }
}

// ---------------- elementwise f32 -> bf16 ----------------
__global__ __launch_bounds__(256) void f32_to_bf16_kernel(
    const float* __restrict__ in, bf16_t* __restrict__ out, int n) {
  int i = blockIdx.x * 256 + threadIdx.x;
  if (i < n) out[i] = (bf16_t)in[i];
}

// ---------------- row softmax over N_ (in place, f32) ----------------
__global__ __launch_bounds__(256) void softmax_rows(float* __restrict__ attn) {
  size_t row = blockIdx.x;
  float* p = attn + row * (size_t)N_;
  __shared__ float sred[8];
  int lane = threadIdx.x & 31, wid = threadIdx.x >> 5;

  float mx = -3.4e38f;
  for (int i = threadIdx.x; i < N_; i += 256) mx = fmaxf(mx, p[i]);
  for (int o = 16; o; o >>= 1) mx = fmaxf(mx, __shfl_xor(mx, o, 32));
  if (lane == 0) sred[wid] = mx;
  __syncthreads();
  if (wid == 0) {
    float v = (lane < 8) ? sred[lane] : -3.4e38f;
    for (int o = 16; o; o >>= 1) v = fmaxf(v, __shfl_xor(v, o, 32));
    if (lane == 0) sred[0] = v;
  }
  __syncthreads();
  float rowmax = sred[0];
  __syncthreads();

  float s = 0.0f;
  for (int i = threadIdx.x; i < N_; i += 256) {
    float e = expf(p[i] - rowmax);
    p[i] = e;
    s += e;
  }
  for (int o = 16; o; o >>= 1) s += __shfl_xor(s, o, 32);
  if (lane == 0) sred[wid] = s;
  __syncthreads();
  if (wid == 0) {
    float v = (lane < 8) ? sred[lane] : 0.0f;
    for (int o = 16; o; o >>= 1) v += __shfl_xor(v, o, 32);
    if (lane == 0) sred[0] = v;
  }
  __syncthreads();
  float inv = 1.0f / sred[0];
  for (int i = threadIdx.x; i < N_; i += 256) p[i] *= inv;
}

// ---------------- LayerNorm over D_, output bf16 ----------------
__global__ __launch_bounds__(256) void layernorm_kernel(
    const float* __restrict__ x, const float* __restrict__ g,
    const float* __restrict__ b, bf16_t* __restrict__ h) {
  int row = blockIdx.x;
  const float* xr = x + (size_t)row * D_;
  bf16_t* hr = h + (size_t)row * D_;
  int lane = threadIdx.x & 31, wid = threadIdx.x >> 5;

  float s = 0.0f, s2 = 0.0f;
  for (int i = threadIdx.x; i < D_; i += 256) {
    float v = xr[i];
    s += v;
    s2 += v * v;
  }
  for (int o = 16; o; o >>= 1) {
    s  += __shfl_xor(s,  o, 32);
    s2 += __shfl_xor(s2, o, 32);
  }
  __shared__ float as[8], as2[8], smu, srs;
  if (lane == 0) { as[wid] = s; as2[wid] = s2; }
  __syncthreads();
  if (threadIdx.x == 0) {
    float S = 0.0f, S2 = 0.0f;
    for (int i = 0; i < 8; ++i) { S += as[i]; S2 += as2[i]; }
    float mu  = S / (float)D_;
    float var = S2 / (float)D_ - mu * mu;
    smu = mu;
    srs = rsqrtf(var + 1e-5f);
  }
  __syncthreads();
  float mu = smu, rs = srs;
  for (int i = threadIdx.x; i < D_; i += 256)
    hr[i] = (bf16_t)((xr[i] - mu) * rs * g[i] + b[i]);
}

// ---------------- host orchestration ----------------
extern "C" void kernel_launch(void* const* d_in, const int* in_sizes, int n_in,
                              void* d_out, int out_size, void* d_ws, size_t ws_size,
                              hipStream_t stream) {
  (void)in_sizes; (void)n_in; (void)out_size; (void)ws_size;

  const float* x    = (const float*)d_in[0];
  const float* q    = (const float*)d_in[1];
  const float* w_q  = (const float*)d_in[2];
  const float* b_q  = (const float*)d_in[3];
  const float* w_k  = (const float*)d_in[4];
  const float* b_k  = (const float*)d_in[5];
  const float* w_v  = (const float*)d_in[6];
  const float* b_v  = (const float*)d_in[7];
  const float* w_o  = (const float*)d_in[8];
  const float* b_o  = (const float*)d_in[9];
  const float* ln_g = (const float*)d_in[10];
  const float* ln_b = (const float*)d_in[11];
  const float* w1   = (const float*)d_in[12];
  const float* b1   = (const float*)d_in[13];
  const float* w2   = (const float*)d_in[14];
  const float* b2   = (const float*)d_in[15];

  float* out_f = (float*)d_out;                        // [B,M,D]
  float* attn  = out_f + (size_t)B_ * M_ * D_;         // [H,B,M,N]

  // workspace carve-up (256B aligned)
  char* wp = (char*)d_ws;
  auto carve = [&](size_t elems, size_t esz) {
    void* r = (void*)wp;
    wp += (elems * esz + 255) & ~(size_t)255;
    return r;
  };
  bf16_t* xbf   = (bf16_t*)carve((size_t)B_ * N_ * D_,   2);
  bf16_t* qbf   = (bf16_t*)carve((size_t)B_ * M_ * D_,   2);
  bf16_t* wqbf  = (bf16_t*)carve((size_t)D_ * D_,        2);
  bf16_t* wkbf  = (bf16_t*)carve((size_t)D_ * D_,        2);
  bf16_t* wvbf  = (bf16_t*)carve((size_t)D_ * D_,        2);
  bf16_t* wobf  = (bf16_t*)carve((size_t)D_ * D_,        2);
  bf16_t* w1bf  = (bf16_t*)carve((size_t)MLP_ * D_,      2);
  bf16_t* w2bf  = (bf16_t*)carve((size_t)D_ * MLP_,      2);
  bf16_t* qpbf  = (bf16_t*)carve((size_t)B_ * M_ * D_,   2); // [B,M,D]
  bf16_t* kpbf  = (bf16_t*)carve((size_t)B_ * N_ * D_,   2); // [B,N,D]
  bf16_t* vptbf = (bf16_t*)carve((size_t)B_ * D_ * N_,   2); // [B,H*HD,N] transposed
  bf16_t* ctxbf = (bf16_t*)carve((size_t)B_ * M_ * D_,   2); // [B,M,D]
  float*  resid = (float*) carve((size_t)B_ * M_ * D_,   4); // [B,M,D]
  bf16_t* hbf   = (bf16_t*)carve((size_t)B_ * M_ * D_,   2);
  bf16_t* gbf   = (bf16_t*)carve((size_t)B_ * M_ * MLP_, 2); // [B,M,MLP]

  auto conv = [&](const float* src, bf16_t* dst, int n) {
    f32_to_bf16_kernel<<<(n + 255) / 256, 256, 0, stream>>>(src, dst, n);
  };
  conv(x,   xbf,  B_ * N_ * D_);
  conv(q,   qbf,  B_ * M_ * D_);
  conv(w_q, wqbf, D_ * D_);
  conv(w_k, wkbf, D_ * D_);
  conv(w_v, wvbf, D_ * D_);
  conv(w_o, wobf, D_ * D_);
  conv(w1,  w1bf, MLP_ * D_);
  conv(w2,  w2bf, D_ * MLP_);

  // Q projection: [4096,1024] x w_qᵀ -> qp bf16
  gemm_wmma<EPI_BF16_BIAS, false><<<dim3(32, 8, 1), 128, 0, stream>>>(
      qbf, 0, 0, D_, wqbf, 0, 0, D_, qpbf, 0, 0, D_,
      b_q, nullptr, 0, B_ * M_, D_, D_, 1, 1.0f);

  // K projection: [8192,1024] x w_kᵀ -> kp bf16
  gemm_wmma<EPI_BF16_BIAS, false><<<dim3(64, 8, 1), 128, 0, stream>>>(
      xbf, 0, 0, D_, wkbf, 0, 0, D_, kpbf, 0, 0, D_,
      b_k, nullptr, 0, B_ * N_, D_, D_, 1, 1.0f);

  // V projection, stored transposed as [B, e, n] so ctx-GEMM B-operand is K-contiguous
  gemm_wmma<EPI_BF16_TRANS, false><<<dim3(16, 8, 4), 128, 0, stream>>>(
      xbf, (size_t)N_ * D_, 0, D_, wvbf, 0, 0, D_,
      vptbf, (size_t)D_ * N_, 0, N_,
      b_v, nullptr, 0, N_, D_, D_, 1, 1.0f);

  // scores = qp·kpᵀ / sqrt(HD), written straight into attn[h,b,m,n]
  gemm_wmma<EPI_SCALE_F32, false><<<dim3(8, 16, B_ * H_), 128, 0, stream>>>(
      qpbf, (size_t)M_ * D_, (size_t)HD_, D_,
      kpbf, (size_t)N_ * D_, (size_t)HD_, D_,
      attn, (size_t)M_ * N_, (size_t)B_ * M_ * N_, N_,
      nullptr, nullptr, 0, M_, N_, HD_, H_, 0.125f);

  // softmax over n, in place in d_out
  softmax_rows<<<B_ * H_ * M_, 256, 0, stream>>>(attn);

  // ctx = attn·vp (A is f32 attn converted inline), stored as ctx[b,m,h*64+hd] bf16
  gemm_wmma<EPI_BF16_BIAS, true><<<dim3(8, 1, B_ * H_), 128, 0, stream>>>(
      attn, (size_t)M_ * N_, (size_t)B_ * M_ * N_, N_,
      vptbf, (size_t)D_ * N_, (size_t)HD_ * N_, N_,
      ctxbf, (size_t)M_ * D_, (size_t)HD_, D_,
      nullptr, nullptr, 0, M_, HD_, N_, H_, 1.0f);

  // O projection + residual (q): resid = ctx·w_oᵀ + b_o + q
  gemm_wmma<EPI_F32_BIAS_ADD, false><<<dim3(32, 8, 1), 128, 0, stream>>>(
      ctxbf, 0, 0, D_, wobf, 0, 0, D_, resid, 0, 0, D_,
      b_o, q, 0, B_ * M_, D_, D_, 1, 1.0f);

  // LayerNorm -> h bf16
  layernorm_kernel<<<B_ * M_, 256, 0, stream>>>(resid, ln_g, ln_b, hbf);

  // MLP up + exact GELU -> g bf16
  gemm_wmma<EPI_GELU_BF16, false><<<dim3(32, 32, 1), 128, 0, stream>>>(
      hbf, 0, 0, D_, w1bf, 0, 0, D_, gbf, 0, 0, MLP_,
      b1, nullptr, 0, B_ * M_, MLP_, D_, 1, 1.0f);

  // MLP down + residual -> final out (f32, d_out)
  gemm_wmma<EPI_F32_BIAS_ADD, false><<<dim3(32, 8, 1), 128, 0, stream>>>(
      gbf, 0, 0, MLP_, w2bf, 0, 0, MLP_, out_f, 0, 0, D_,
      b2, resid, 0, B_ * M_, D_, MLP_, 1, 1.0f);
}
